// TRNN_34677565948680
// MI455X (gfx1250) — matrix operand order repaired and checked by reading, not compiled, not run on hardware
//
#include <hip/hip_runtime.h>

// MI455X fused TRNN: 128-step relu-RNN (bf16 WMMA 16x16x32) + fused time
// projection (f32 WMMA 16x16x4), hs never touches HBM. One WG = 4 waves =
// 16 batch rows; wave w owns h-slice [16w,16w+16). ~600MB HBM traffic total
// (read ms once, write q once) -> memory-bound at ~26us on 23.3 TB/s.
//
// Round-2 refinements: two independent WMMA accumulation chains (m-path and
// h-path) interleave to hide WMMA->WMMA hazard NOPs; ReLU via v_med3_f32.

typedef __attribute__((ext_vector_type(16))) __bf16 v16bf;
typedef __attribute__((ext_vector_type(8)))  __bf16 v8bf;
typedef __attribute__((ext_vector_type(8)))  float  v8f;
typedef __attribute__((ext_vector_type(2)))  float  v2f;

#define T_DIM 128
#define B_DIM 16384
#define H_DIM 64
#define P_DIM 16

// Pack 8+8 contiguous f32 into a bf16 A/B operand lane vector.
__device__ __forceinline__ v16bf cvt16(const float* p0, const float* p1) {
  float4 f0 = *(const float4*)(p0);
  float4 f1 = *(const float4*)(p0 + 4);
  float4 f2 = *(const float4*)(p1);
  float4 f3 = *(const float4*)(p1 + 4);
  v16bf a;
  a[0]=(__bf16)f0.x;  a[1]=(__bf16)f0.y;  a[2]=(__bf16)f0.z;  a[3]=(__bf16)f0.w;
  a[4]=(__bf16)f1.x;  a[5]=(__bf16)f1.y;  a[6]=(__bf16)f1.z;  a[7]=(__bf16)f1.w;
  a[8]=(__bf16)f2.x;  a[9]=(__bf16)f2.y;  a[10]=(__bf16)f2.z; a[11]=(__bf16)f2.w;
  a[12]=(__bf16)f3.x; a[13]=(__bf16)f3.y; a[14]=(__bf16)f3.z; a[15]=(__bf16)f3.w;
  return a;
}

// Gather 8+8 contiguous bf16 (16B-aligned) from LDS into a lane vector.
__device__ __forceinline__ v16bf ld16bf(const __bf16* p0, const __bf16* p1) {
  v8bf u0 = *(const v8bf*)p0;
  v8bf u1 = *(const v8bf*)p1;
  v16bf a;
#pragma unroll
  for (int i = 0; i < 8; ++i) { a[i] = u0[i]; a[8 + i] = u1[i]; }
  return a;
}

__global__ __launch_bounds__(128) void trnn_fused(
    const float* __restrict__ ms,    const float* __restrict__ W_rs,
    const float* __restrict__ b_rs,  const float* __restrict__ W_vs,
    const float* __restrict__ b_vs,  const float* __restrict__ tp_bs,
    const float* __restrict__ W_proj,const float* __restrict__ b_proj,
    float* __restrict__ out)
{
  // h state (bf16, ping-pong) + 4-step f32 ring buffer for the projection.
  __shared__ __align__(16) __bf16 hbuf[2][16 * H_DIM];   // 4 KB
  __shared__ __align__(16) float  pbuf[4][16 * H_DIM];   // 16 KB

  const int tid   = threadIdx.x;
  const int lane  = tid & 31;
  const int wave  = tid >> 5;      // 0..3 -> h-slice
  const int hi16  = lane >> 4;     // lane half (0/1)
  const int l16   = lane & 15;
  const int slice = wave * 16;     // this wave's output-h base
  const int bg    = blockIdx.x * 16;

  // ---- stage weight B-tiles (B[k,n] = W[slice+n][k], bf16) in VGPRs ----
  // B 32x16 bf16 layout: lane n=l16; lane half picks K runs
  // {r0..r0+7, r0+16..r0+23}, r0 = kc + 8*hi16 (matches A).
  v16bf wrsB[2], wvsB[2];
#pragma unroll
  for (int c = 0; c < 2; ++c) {
    const int kbase = hi16 * 8 + c * 32;
    const float* prs = W_rs + (slice + l16) * H_DIM + kbase;
    const float* pvs = W_vs + (slice + l16) * H_DIM + kbase;
    wrsB[c] = cvt16(prs, prs + 16);
    wvsB[c] = cvt16(pvs, pvs + 16);
  }

  // Recurrence C init: bias depends only on N=h (lane), same for all rows.
  const float biasv = b_rs[slice + l16] + b_vs[slice + l16] + tp_bs[slice + l16];
  v8f cbias, czero;
#pragma unroll
  for (int r = 0; r < 8; ++r) { cbias[r] = biasv; czero[r] = 0.0f; }

  // Projection accumulators: q[bi] is D[p,h] (p = r + 8*hi16, h = slice+l16).
  v8f q[16];
#pragma unroll
  for (int bi = 0; bi < 16; ++bi)
#pragma unroll
    for (int r = 0; r < 8; ++r) q[bi][r] = b_proj[hi16 * 8 + r];

  // h0 = 0
  for (int i = tid; i < 2 * 16 * H_DIM; i += 128) (&hbuf[0][0])[i] = (__bf16)0.0f;
  __syncthreads();

  const float* mrow0 = ms + (size_t)(bg + l16) * H_DIM;  // A rows: M = l16

  for (int t = 0; t < T_DIM; ++t) {
    const int cur = t & 1, nxt = cur ^ 1, slot = t & 3;

    // ---- A-tiles: m_t from HBM (f32 -> bf16), layout-exact per lane ----
    // lane half picks K runs {r0..r0+7, r0+16..r0+23}, r0 = 8*hi16 + 32c.
    const float* mrow = mrow0 + (size_t)t * (B_DIM * H_DIM);
    v16bf mA[2], hA[2];
#pragma unroll
    for (int c = 0; c < 2; ++c) {
      const int r0 = hi16 * 8 + c * 32;
      mA[c] = cvt16(mrow + r0, mrow + r0 + 16);
      const __bf16* hrow = &hbuf[cur][l16 * H_DIM];
      hA[c] = ld16bf(hrow + r0, hrow + r0 + 16);
    }
    if (t + 1 < T_DIM)  // uniform branch: prefetch next step's ms tile
      __builtin_prefetch(mrow + (size_t)(B_DIM * H_DIM), 0, 3);

    // ---- h_new = relu(m@Wrs^T + h@Wvs^T + bias) ----
    // Two independent 2-deep WMMA chains interleave, hiding D->C hazards.
    v8f accm = cbias, acch = czero;
    accm = __builtin_amdgcn_wmma_f32_16x16x32_bf16(false, mA[0], false, wrsB[0],
                                                   (short)0, accm, false, false);
    acch = __builtin_amdgcn_wmma_f32_16x16x32_bf16(false, hA[0], false, wvsB[0],
                                                   (short)0, acch, false, false);
    accm = __builtin_amdgcn_wmma_f32_16x16x32_bf16(false, mA[1], false, wrsB[1],
                                                   (short)0, accm, false, false);
    acch = __builtin_amdgcn_wmma_f32_16x16x32_bf16(false, hA[1], false, wvsB[1],
                                                   (short)0, acch, false, false);
    v8f acc;
#pragma unroll
    for (int r = 0; r < 8; ++r)  // relu(x) = med3(x, 0, +inf): one v_med3_f32
      acc[r] = __builtin_amdgcn_fmed3f(accm[r] + acch[r], 0.0f,
                                       __builtin_inff());

    // ---- scatter D (row b = r + 8*hi16, col h = slice+l16) to LDS ----
#pragma unroll
    for (int r = 0; r < 8; ++r) {
      const int b = r + hi16 * 8, h = slice + l16;
      hbuf[nxt][b * H_DIM + h] = (__bf16)acc[r];
      pbuf[slot][b * H_DIM + h] = acc[r];
    }

    // ---- every 4 steps: q[b,p,h] += W_proj[p,t0..t0+3] x hs (f32 WMMA) ----
    // Reads only this wave's own pbuf writes -> same-wave DS order, no barrier.
    if ((t & 3) == 3) {
      const int t0 = t - 3;
      v2f pA;  // A 16x4 f32: VGPR0 K=2*hi16, VGPR1 K=2*hi16+1, M=l16
      pA[0] = W_proj[l16 * T_DIM + t0 + 2 * hi16];
      pA[1] = W_proj[l16 * T_DIM + t0 + 2 * hi16 + 1];
#pragma unroll
      for (int bi = 0; bi < 16; ++bi) {
        v2f pB;  // B 4x16 f32: VGPR0 K=2*hi16, VGPR1 K=2*hi16+1, N=l16
        pB[0] = pbuf[2 * hi16 + 0][bi * H_DIM + slice + l16];
        pB[1] = pbuf[2 * hi16 + 1][bi * H_DIM + slice + l16];
        q[bi] = __builtin_amdgcn_wmma_f32_16x16x4_f32(false, pA, false, pB,
                                                      (short)0, q[bi],
                                                      false, false);
      }
    }
    __syncthreads();  // h_new visible to all waves before next step's reads
  }

  // ---- write q[b,p,h]: D layout -> out[(bg+bi)*P*H + p*H + h] ----
#pragma unroll
  for (int bi = 0; bi < 16; ++bi) {
    const size_t base = (size_t)(bg + bi) * (P_DIM * H_DIM);
#pragma unroll
    for (int r = 0; r < 8; ++r) {
      const int p = r + hi16 * 8;
      out[base + p * H_DIM + slice + l16] = q[bi][r];
    }
  }
}

extern "C" void kernel_launch(void* const* d_in, const int* in_sizes, int n_in,
                              void* d_out, int out_size, void* d_ws, size_t ws_size,
                              hipStream_t stream) {
  const float* ms     = (const float*)d_in[0];
  const float* W_rs   = (const float*)d_in[1];
  const float* b_rs   = (const float*)d_in[2];
  const float* W_vs   = (const float*)d_in[3];
  const float* b_vs   = (const float*)d_in[4];
  const float* tp_bs  = (const float*)d_in[5];
  const float* W_proj = (const float*)d_in[6];
  const float* b_proj = (const float*)d_in[7];
  float* out = (float*)d_out;
  (void)in_sizes; (void)n_in; (void)out_size; (void)d_ws; (void)ws_size;

  dim3 grid(B_DIM / 16), block(128);
  trnn_fused<<<grid, block, 0, stream>>>(ms, W_rs, b_rs, W_vs, b_vs, tp_bs,
                                         W_proj, b_proj, out);
}
